// MambaPOD_space_54795192762430
// MI455X (gfx1250) — matrix operand order repaired and checked by reading, not compiled, not run on hardware
//
#include <hip/hip_runtime.h>
#include <cstdint>
#include <cstddef>

typedef __attribute__((ext_vector_type(16))) _Float16 v16h;
typedef __attribute__((ext_vector_type(8)))  float    v8f;

union FragAB { v16h v; uint32_t u[8]; };

// ---------------------------------------------------------------------------
// Problem constants (match setup_inputs)
// ---------------------------------------------------------------------------
#define BSZ   4
#define SEQL  2048
#define DMOD  192
#define DINN  384
#define DST   64
#define DCONV 4
#define DRANK 12
#define BL    (BSZ * SEQL)          // 8192 tokens
#define NPAD2 80                    // DR+DS = 76 padded to 80

// ---------------------------------------------------------------------------
// Pack weight W [Nreal x K] (row-major, f32) into WMMA B-fragment layout:
// Bp[(ntile*kchunks + kc)*32 + lane][16 halves], element e holds
// B[k = kc*32 + (lane>>4)*16 + e][n = ntile*16 + (lane&15)] = W[n*K + k].
// Zero-pads n >= Nreal.
// ---------------------------------------------------------------------------
__global__ void pack_w_kernel(const float* __restrict__ W, _Float16* __restrict__ Bp,
                              int Nreal, int K, int ntiles) {
  const int kchunks = K / 32;
  const int t = blockIdx.x * 256 + threadIdx.x;
  if (t >= ntiles * kchunks * 32) return;
  const int lane = t & 31;
  const int kc   = (t >> 5) % kchunks;
  const int nt   = t / (32 * kchunks);
  const int n    = nt * 16 + (lane & 15);
  const int kb   = kc * 32 + (lane >> 4) * 16;
  _Float16* dst = Bp + (size_t)t * 16;
#pragma unroll
  for (int e = 0; e < 16; ++e) {
    float v = (n < Nreal) ? W[(size_t)n * K + kb + e] : 0.0f;
    dst[e] = (_Float16)v;
  }
}

// ---------------------------------------------------------------------------
// Fused residual add + LayerNorm. One 64-thread block per token row.
// Writes f32 residual (second output) and f16 normalized h (GEMM1 input).
// ---------------------------------------------------------------------------
__global__ void resid_ln_kernel(const float* __restrict__ hs, const float* __restrict__ res_in,
                                const float* __restrict__ ln_w, const float* __restrict__ ln_b,
                                float* __restrict__ res_out, _Float16* __restrict__ h16) {
  __shared__ float sbuf[64];
  const int row = blockIdx.x;
  const int tid = threadIdx.x;
  const size_t base = (size_t)row * DMOD;

  float v[3];
  float s = 0.0f;
#pragma unroll
  for (int i = 0; i < 3; ++i) {
    const int c = tid + i * 64;
    float x = res_in[base + c] + hs[base + c];
    v[i] = x; s += x;
    res_out[base + c] = x;
  }
  sbuf[tid] = s;
  for (int off = 32; off > 0; off >>= 1) { __syncthreads(); if (tid < off) sbuf[tid] += sbuf[tid + off]; }
  __syncthreads();
  const float mu = sbuf[0] * (1.0f / DMOD);
  __syncthreads();

  float s2 = 0.0f;
#pragma unroll
  for (int i = 0; i < 3; ++i) { float d = v[i] - mu; s2 += d * d; }
  sbuf[tid] = s2;
  for (int off = 32; off > 0; off >>= 1) { __syncthreads(); if (tid < off) sbuf[tid] += sbuf[tid + off]; }
  __syncthreads();
  const float inv = rsqrtf(sbuf[0] * (1.0f / DMOD) + 1e-5f);

#pragma unroll
  for (int i = 0; i < 3; ++i) {
    const int c = tid + i * 64;
    float h = (v[i] - mu) * inv * ln_w[c] + ln_b[c];
    h16[base + c] = (_Float16)h;
  }
}

// ---------------------------------------------------------------------------
// Generic wave-tile WMMA GEMM: C[M x N] = A[M x K] (f16 row-major) * Bpacked.
// One wave per 16x16 output tile; 4 waves per block. N multiple of 16,
// K multiple of 32, M multiple of 16. Grid must cover Mtiles*Ntiles exactly.
// ---------------------------------------------------------------------------
__global__ __launch_bounds__(128) void gemm_wmma_kernel(
    const _Float16* __restrict__ A, const _Float16* __restrict__ Bp,
    float* __restrict__ C, int K, int N, int Mtiles, int kchunks) {
  const int wave = threadIdx.x >> 5;
  const int lane = threadIdx.x & 31;
  const int t     = blockIdx.x * 4 + wave;
  const int tileM = t % Mtiles;
  const int tileN = t / Mtiles;
  const int r    = lane & 15;
  const int half = lane >> 4;

  v8f acc = {};
  const _Float16* arow = A + (size_t)(tileM * 16 + r) * K;
  const _Float16* bbase = Bp + ((size_t)tileN * kchunks * 32 + lane) * 16;

  for (int kc = 0; kc < kchunks; ++kc) {
    FragAB a, b;
    // A fragment: contiguous K order matching packed B: lanes 0-15 K 0..15,
    // lanes 16-31 K 16..31.
    const uint32_t* ap = (const uint32_t*)(arow + kc * 32);
#pragma unroll
    for (int vg = 0; vg < 8; ++vg) {
      const int pair = vg + half * 8;
      a.u[vg] = ap[pair];
    }
    b.v = *(const v16h*)(bbase + (size_t)kc * 32 * 16);
    acc = __builtin_amdgcn_wmma_f32_16x16x32_f16(
        false, a.v, false, b.v, (short)0, acc, false, false);
  }
  // D layout: VGPR vg -> M = tileM*16 + half*8 + vg, N = tileN*16 + r
  float* crow = C + (size_t)(tileM * 16 + half * 8) * N + tileN * 16 + r;
#pragma unroll
  for (int vg = 0; vg < 8; ++vg) crow[(size_t)vg * N] = acc[vg];
}

// ---------------------------------------------------------------------------
// Depthwise causal conv(4) + bias + SiLU. x comes from xz[..., :DINN].
// ---------------------------------------------------------------------------
__global__ void conv_silu_kernel(const float* __restrict__ xz, const float* __restrict__ conv_w,
                                 const float* __restrict__ conv_b,
                                 float* __restrict__ x32, _Float16* __restrict__ x16) {
  const int idx = blockIdx.x * 256 + threadIdx.x;
  if (idx >= BL * DINN) return;
  const int c   = idx % DINN;
  const int row = idx / DINN;
  const int l   = row % SEQL;
  const int b   = row / SEQL;
  float acc = conv_b[c];
#pragma unroll
  for (int j = 0; j < DCONV; ++j) {
    const int ls = l - (DCONV - 1) + j;
    if (ls >= 0)
      acc += conv_w[c * DCONV + j] * xz[((size_t)b * SEQL + ls) * (2 * DINN) + c];
  }
  const float y = acc / (1.0f + __expf(-acc));   // silu
  x32[idx] = y;
  x16[idx] = (_Float16)y;
}

// ---------------------------------------------------------------------------
// dt = softplus(xdb[:, :DRANK] @ W_dt^T + b_dt)   (K=12, scalar VALU)
// ---------------------------------------------------------------------------
__global__ void dt_kernel(const float* __restrict__ xdb, const float* __restrict__ W_dt,
                          const float* __restrict__ b_dt, float* __restrict__ dt) {
  const int idx = blockIdx.x * 256 + threadIdx.x;
  if (idx >= BL * DINN) return;
  const int d = idx % DINN;
  const int m = idx / DINN;
  float acc = b_dt[d];
  const float* xr = xdb + (size_t)m * NPAD2;
  const float* wr = W_dt + (size_t)d * DRANK;
#pragma unroll
  for (int r = 0; r < DRANK; ++r) acc += xr[r] * wr[r];
  dt[idx] = (acc > 20.0f) ? acc : log1pf(__expf(acc));
}

// ---------------------------------------------------------------------------
// Selective scan. One wave32 per (batch, channel): 64 states = 2 per lane.
// Per step: h = exp(dt*A)*h + (dt*u)*B_t ; y = sum_n h*C[n] (shfl_xor reduce)
// ---------------------------------------------------------------------------
__global__ __launch_bounds__(128) void scan_kernel(
    const float* __restrict__ dt, const float* __restrict__ x32,
    const float* __restrict__ xdb, const float* __restrict__ A_log,
    const float* __restrict__ C_fixed, float* __restrict__ y) {
  const int wave = threadIdx.x >> 5;
  const int lane = threadIdx.x & 31;
  const int w = blockIdx.x * 4 + wave;      // 0 .. BSZ*DINN-1
  const int b = w / DINN;
  const int d = w % DINN;
  const int n0 = lane, n1 = lane + 32;

  const float A0 = -__expf(A_log[d * DST + n0]);
  const float A1 = -__expf(A_log[d * DST + n1]);
  const float C0 = C_fixed[n0];
  const float C1 = C_fixed[n1];

  float h0 = 0.0f, h1 = 0.0f;
  const size_t tok0 = (size_t)b * SEQL;
  for (int l = 0; l < SEQL; ++l) {
    const size_t row = tok0 + l;
    const float dtv = dt[row * DINN + d];
    const float u   = x32[row * DINN + d];
    const float du  = dtv * u;
    const float b0  = xdb[row * NPAD2 + DRANK + n0];
    const float b1  = xdb[row * NPAD2 + DRANK + n1];
    h0 = __expf(dtv * A0) * h0 + du * b0;
    h1 = __expf(dtv * A1) * h1 + du * b1;
    float part = h0 * C0 + h1 * C1;
#pragma unroll
    for (int off = 16; off > 0; off >>= 1) part += __shfl_xor(part, off);
    if (lane == 0) y[row * DINN + d] = part;
  }
}

// ---------------------------------------------------------------------------
// y = (y_scan + x*D_skip) * silu(z)  ->  f16 for GEMM3
// ---------------------------------------------------------------------------
__global__ void gate_kernel(const float* __restrict__ y_scan, const float* __restrict__ x32,
                            const float* __restrict__ xz, const float* __restrict__ D_skip,
                            _Float16* __restrict__ y16) {
  const int idx = blockIdx.x * 256 + threadIdx.x;
  if (idx >= BL * DINN) return;
  const int d = idx % DINN;
  const int m = idx / DINN;
  const float z = xz[(size_t)m * (2 * DINN) + DINN + d];
  const float g = z / (1.0f + __expf(-z));
  const float yy = (y_scan[idx] + x32[idx] * D_skip[d]) * g;
  y16[idx] = (_Float16)yy;
}

// ---------------------------------------------------------------------------
// Host orchestration
// ---------------------------------------------------------------------------
extern "C" void kernel_launch(void* const* d_in, const int* in_sizes, int n_in,
                              void* d_out, int out_size, void* d_ws, size_t ws_size,
                              hipStream_t stream) {
  (void)in_sizes; (void)n_in; (void)out_size; (void)ws_size;
  const float* hs      = (const float*)d_in[0];
  const float* res_in  = (const float*)d_in[1];
  const float* ln_w    = (const float*)d_in[2];
  const float* ln_b    = (const float*)d_in[3];
  const float* W_in    = (const float*)d_in[4];
  const float* conv_w  = (const float*)d_in[5];
  const float* conv_b  = (const float*)d_in[6];
  const float* W_xproj = (const float*)d_in[7];
  const float* W_dt    = (const float*)d_in[8];
  const float* b_dt    = (const float*)d_in[9];
  const float* A_log   = (const float*)d_in[10];
  const float* D_skip  = (const float*)d_in[11];
  const float* C_fixed = (const float*)d_in[12];
  const float* W_out   = (const float*)d_in[13];

  float* out     = (float*)d_out;                 // (B,L,D) flat
  float* res_out = out + (size_t)BL * DMOD;       // second tuple element

  // Carve 256B-aligned workspace regions
  char* wsp = (char*)d_ws;
  auto carve = [&](size_t bytes) -> char* {
    char* p = wsp; wsp += (bytes + 255) & ~(size_t)255; return p;
  };
  _Float16* h16  = (_Float16*)carve((size_t)BL * DMOD * 2);
  float*    xz   = (float*)   carve((size_t)BL * 2 * DINN * 4);
  float*    x32  = (float*)   carve((size_t)BL * DINN * 4);
  _Float16* x16  = (_Float16*)carve((size_t)BL * DINN * 2);
  float*    xdb  = (float*)   carve((size_t)BL * NPAD2 * 4);
  float*    dtb  = (float*)   carve((size_t)BL * DINN * 4);
  float*    ysc  = (float*)   carve((size_t)BL * DINN * 4);
  _Float16* y16  = (_Float16*)carve((size_t)BL * DINN * 2);
  _Float16* bp1  = (_Float16*)carve((size_t)48 * 6  * 32 * 16 * 2);  // W_in packed
  _Float16* bp2  = (_Float16*)carve((size_t)5  * 12 * 32 * 16 * 2);  // W_xproj packed (N padded 80)
  _Float16* bp3  = (_Float16*)carve((size_t)12 * 12 * 32 * 16 * 2);  // W_out packed

  // --- weight packing (independent, L2-resident) ---
  pack_w_kernel<<<(48 * 6 * 32 + 255) / 256, 256, 0, stream>>>(W_in,    bp1, 2 * DINN,        DMOD, 48);
  pack_w_kernel<<<(5  * 12 * 32 + 255) / 256, 256, 0, stream>>>(W_xproj, bp2, DRANK + DST,     DINN, 5);
  pack_w_kernel<<<(12 * 12 * 32 + 255) / 256, 256, 0, stream>>>(W_out,   bp3, DMOD,            DINN, 12);

  // --- residual + LayerNorm ---
  resid_ln_kernel<<<BL, 64, 0, stream>>>(hs, res_in, ln_w, ln_b, res_out, h16);

  // --- GEMM1: xz = h @ W_in^T  (8192x192 x 192x768) ---
  gemm_wmma_kernel<<<(BL / 16) * 48 / 4, 128, 0, stream>>>(h16, bp1, xz, DMOD, 2 * DINN, BL / 16, DMOD / 32);

  // --- depthwise conv + SiLU ---
  conv_silu_kernel<<<(BL * DINN + 255) / 256, 256, 0, stream>>>(xz, conv_w, conv_b, x32, x16);

  // --- GEMM2: xdb = x @ W_xproj^T  (8192x384 x 384x80pad) ---
  gemm_wmma_kernel<<<(BL / 16) * 5 / 4, 128, 0, stream>>>(x16, bp2, xdb, DINN, NPAD2, BL / 16, DINN / 32);

  // --- dt projection + softplus ---
  dt_kernel<<<(BL * DINN + 255) / 256, 256, 0, stream>>>(xdb, W_dt, b_dt, dtb);

  // --- selective scan: one wave per (b, d) ---
  scan_kernel<<<(BSZ * DINN) / 4, 128, 0, stream>>>(dtb, x32, xdb, A_log, C_fixed, ysc);

  // --- gate ---
  gate_kernel<<<(BL * DINN + 255) / 256, 256, 0, stream>>>(ysc, x32, xz, D_skip, y16);

  // --- GEMM3: out = y @ W_out^T  (8192x384 x 384x192), straight to d_out ---
  gemm_wmma_kernel<<<(BL / 16) * 12 / 4, 128, 0, stream>>>(y16, bp3, out, DINN, DMOD, BL / 16, DINN / 32);
}
